// Net_80942953660684
// MI455X (gfx1250) — matrix operand order repaired and checked by reading, not compiled, not run on hardware
//
#include <hip/hip_runtime.h>

// ---------------------------------------------------------------------------
// Problem constants (from reference): N=8, D_SEM=2048, D_HIE=768, H=W=32,
// P=H*W=1024, NUM_CLS=21 (padded to 32 for WMMA tiles), 2 clusters, 10 iters.
// ---------------------------------------------------------------------------
#define NB 8
#define DS 2048
#define DH 768
#define PP 1024
#define NC 21
#define CP 32

typedef __attribute__((ext_vector_type(16))) _Float16 v16h;
typedef __attribute__((ext_vector_type(8)))  _Float16 v8h;
typedef __attribute__((ext_vector_type(8)))  float    v8f;

__device__ inline v8f wmma16(v16h a, v16h b, v8f c) {
  return __builtin_amdgcn_wmma_f32_16x16x32_f16(false, a, false, b, (short)0, c,
                                                false, false);
}

// A-fragment (16x32 f16): lane<16 -> K {0..15}; lane>=16 -> {16..31}
__device__ inline v16h load_afrag(const _Float16* row, int k0, int hi) {
  return *(const v16h*)(row + k0 + hi * 16);
}

// blockDim.x == 256 everywhere
__device__ inline float blk_sum(float v, float* red) {
  int t = threadIdx.x;
  red[t] = v; __syncthreads();
  for (int s = 128; s > 0; s >>= 1) {
    if (t < s) red[t] += red[t + s];
    __syncthreads();
  }
  float r = red[0]; __syncthreads();
  return r;
}
__device__ inline float blk_max(float v, float* red) {
  int t = threadIdx.x;
  red[t] = v; __syncthreads();
  for (int s = 128; s > 0; s >>= 1) {
    if (t < s) red[t] = fmaxf(red[t], red[t + s]);
    __syncthreads();
  }
  float r = red[0]; __syncthreads();
  return r;
}

// ---------------------------------------------------------------------------
// Per-pixel sum of squares over feature dim:  ss[n][p] = sum_d src[n][d][p]^2
// grid (P/32, N), block 256 = (32 px, 8 d-groups), coalesced over p.
// ---------------------------------------------------------------------------
__global__ void k_sumsq(const float* __restrict__ src, float* __restrict__ ss,
                        int D) {
  int n = blockIdx.y, p0 = blockIdx.x * 32;
  int px = threadIdx.x & 31, dg = threadIdx.x >> 5;
  const float* base = src + (size_t)n * D * PP + p0 + px;
  float acc = 0.f;
  for (int d = dg; d < D; d += 8) {
    float v = base[(size_t)d * PP];
    acc += v * v;
  }
  __shared__ float red[8][33];
  red[dg][px] = acc; __syncthreads();
  if (dg == 0) {
    float s = 0.f;
    for (int g = 0; g < 8; ++g) s += red[g][px];
    ss[n * PP + p0 + px] = s;
  }
}

// ---------------------------------------------------------------------------
// LDS-tiled transpose [n][d][p] -> [n][p][d], l2-normalized over d, f16 out.
// Optionally also writes the raw f32 transpose (X for kmeans).
// grid (D/64, P/32, N).
// ---------------------------------------------------------------------------
__global__ void k_transpose_norm(const float* __restrict__ src,
                                 const float* __restrict__ ss,
                                 _Float16* __restrict__ outH,
                                 float* __restrict__ outRaw, int D, float eps) {
  int d0 = blockIdx.x * 64, p0 = blockIdx.y * 32, n = blockIdx.z;
  __shared__ float t[64][33];
  __shared__ float inv[32];
  int tid = threadIdx.x;
  int px = tid & 31, dg = tid >> 5;
  for (int dd = dg; dd < 64; dd += 8)
    t[dd][px] = src[((size_t)n * D + d0 + dd) * PP + p0 + px];
  if (tid < 32) inv[tid] = 1.f / (sqrtf(ss[n * PP + p0 + tid]) + eps);
  __syncthreads();
  int dx = tid & 63, pg = tid >> 6;
  for (int pp = pg; pp < 32; pp += 4) {
    float v = t[dx][pp];
    size_t o = ((size_t)n * PP + p0 + pp) * D + d0 + dx;
    outH[o] = (_Float16)(v * inv[pp]);
    if (outRaw) outRaw[o] = v;
  }
}

// ---------------------------------------------------------------------------
// cam[n][c][p] = sum_d sem[n][d][p] * w_cls[c][d], c in 0..19.
// w_cls staged through LDS in 128-d chunks; 20 FMA per global load.
// grid (P/64, N), block 256 = (64 px, 4 d-quarters).
// ---------------------------------------------------------------------------
__global__ void __launch_bounds__(256) k_cam(const float* __restrict__ sem,
                                             const float* __restrict__ wcls,
                                             float* __restrict__ cam) {
  int p0 = blockIdx.x * 64, n = blockIdx.y;
  __shared__ float wt[20][128];
  __shared__ float ct[20][64];
  int tid = threadIdx.x;
  int px = tid & 63, dq = tid >> 6;
  for (int i = tid; i < 20 * 64; i += 256) ct[i >> 6][i & 63] = 0.f;
  float acc[20];
#pragma unroll
  for (int c = 0; c < 20; ++c) acc[c] = 0.f;
  for (int d0 = 0; d0 < DS; d0 += 128) {
    __syncthreads();
    for (int i = tid; i < 20 * 128; i += 256)
      wt[i >> 7][i & 127] = wcls[(size_t)(i >> 7) * DS + d0 + (i & 127)];
    __syncthreads();
    for (int dd = dq * 32; dd < dq * 32 + 32; ++dd) {
      float s = sem[((size_t)n * DS + d0 + dd) * PP + p0 + px];
#pragma unroll
      for (int c = 0; c < 20; ++c) acc[c] += s * wt[c][dd];
    }
  }
  __syncthreads();
#pragma unroll
  for (int c = 0; c < 20; ++c) atomicAdd(&ct[c][px], acc[c]);
  __syncthreads();
  for (int i = tid; i < 20 * 64; i += 256)
    cam[((size_t)n * 20 + (i >> 6)) * PP + p0 + (i & 63)] = ct[i >> 6][i & 63];
}

// score[n][c] = mean_p cam ; camMax[n][c] = max_p relu(cam). grid (20, N).
__global__ void k_cam_max_score(const float* __restrict__ cam,
                                float* __restrict__ score,
                                float* __restrict__ camMax) {
  int c = blockIdx.x, n = blockIdx.y;
  const float* row = cam + ((size_t)n * 20 + c) * PP;
  float sm = 0.f, ssum = 0.f;
  for (int p = threadIdx.x; p < PP; p += 256) {
    float v = row[p];
    ssum += v;
    sm = fmaxf(sm, v > 0.f ? v : 0.f);
  }
  __shared__ float red[256];
  float tsum = blk_sum(ssum, red);
  float tmax = blk_max(sm, red);
  if (threadIdx.x == 0) {
    score[n * 20 + c] = tsum * (1.f / (float)PP);
    camMax[n * 20 + c] = tmax;
  }
}

// norm_cam (masked, with bkg channel) -> d_out + padded f16 copy for WMMA B.
// grid (P/256, N).
__global__ void k_normcam(const float* __restrict__ cam,
                          const float* __restrict__ camMax,
                          const float* __restrict__ vmask,
                          float* __restrict__ ncOut,
                          _Float16* __restrict__ nc16) {
  int n = blockIdx.y;
  int p = blockIdx.x * 256 + threadIdx.x;
  float bkgmax = 0.f;
  float nv[20];
#pragma unroll
  for (int c = 0; c < 20; ++c) {
    float v = cam[((size_t)n * 20 + c) * PP + p];
    v = v > 0.f ? v : 0.f;
    v = v / (camMax[n * 20 + c] + 1e-5f);
    nv[c] = v;
    bkgmax = fmaxf(bkgmax, v);
  }
  float o0 = (1.f - bkgmax) * vmask[n * NC + 0];
  ncOut[(size_t)n * NC * PP + p] = o0;
  nc16[(size_t)n * CP * PP + p] = (_Float16)o0;
#pragma unroll
  for (int c = 0; c < 20; ++c) {
    float o = nv[c] * vmask[n * NC + c + 1];
    ncOut[((size_t)n * NC + c + 1) * PP + p] = o;
    nc16[((size_t)n * CP + c + 1) * PP + p] = (_Float16)o;
  }
  for (int c = NC; c < CP; ++c)
    nc16[((size_t)n * CP + c) * PP + p] = (_Float16)0.f;
}

// camsum[n][c] = sum_p norm_cam.  grid (21, N).
__global__ void k_camsum(const float* __restrict__ ncOut,
                         float* __restrict__ camsum) {
  int c = blockIdx.x, n = blockIdx.y;
  const float* row = ncOut + ((size_t)n * NC + c) * PP;
  float s = 0.f;
  for (int p = threadIdx.x; p < PP; p += 256) s += row[p];
  __shared__ float red[256];
  float t = blk_sum(s, red);
  if (threadIdx.x == 0) camsum[n * NC + c] = t;
}

// ---------------------------------------------------------------------------
// corr[n] = relu(fs^T fs): (1024x2048)x(2048x1024) in f16 WMMA, f32 accum.
// Each wave: one 32x64 tile (2 i-tiles x 4 j-tiles, 8 accumulators) with a
// 1-deep software pipeline: fragment loads for k+32 are issued before the 8
// WMMAs of step k so the s_wait for them lands after a full WMMA burst.
// 12 b128 loads per 8 WMMAs. grid (16, 4, N).
// ---------------------------------------------------------------------------
__global__ void __launch_bounds__(256) k_corr_wmma(
    const _Float16* __restrict__ fsT, _Float16* __restrict__ corr16) {
  int n = blockIdx.z;
  int wave = threadIdx.x >> 5, lane = threadIdx.x & 31;
  int l15 = lane & 15, hi = lane >> 4;
  int i0 = (blockIdx.y * 8 + wave) * 32;
  int j0 = blockIdx.x * 64;
  const _Float16* A0 = fsT + (size_t)(n * PP + i0 + l15) * DS;
  const _Float16* A1 = A0 + (size_t)16 * DS;
  const _Float16* B = fsT + (size_t)(n * PP + j0 + l15) * DS + hi * 16;
  v8f c00 = {}, c01 = {}, c02 = {}, c03 = {};
  v8f c10 = {}, c11 = {}, c12 = {}, c13 = {};
  v16h a0 = load_afrag(A0, 0, hi);
  v16h a1 = load_afrag(A1, 0, hi);
  v16h b0 = *(const v16h*)(B);
  v16h b1 = *(const v16h*)(B + (size_t)16 * DS);
  v16h b2 = *(const v16h*)(B + (size_t)32 * DS);
  v16h b3 = *(const v16h*)(B + (size_t)48 * DS);
  for (int k0 = 32; k0 < DS; k0 += 32) {
    v16h na0 = load_afrag(A0, k0, hi);
    v16h na1 = load_afrag(A1, k0, hi);
    const _Float16* pb = B + k0;
    v16h nb0 = *(const v16h*)(pb);
    v16h nb1 = *(const v16h*)(pb + (size_t)16 * DS);
    v16h nb2 = *(const v16h*)(pb + (size_t)32 * DS);
    v16h nb3 = *(const v16h*)(pb + (size_t)48 * DS);
    c00 = wmma16(a0, b0, c00);
    c01 = wmma16(a0, b1, c01);
    c02 = wmma16(a0, b2, c02);
    c03 = wmma16(a0, b3, c03);
    c10 = wmma16(a1, b0, c10);
    c11 = wmma16(a1, b1, c11);
    c12 = wmma16(a1, b2, c12);
    c13 = wmma16(a1, b3, c13);
    a0 = na0; a1 = na1;
    b0 = nb0; b1 = nb1; b2 = nb2; b3 = nb3;
  }
  c00 = wmma16(a0, b0, c00);
  c01 = wmma16(a0, b1, c01);
  c02 = wmma16(a0, b2, c02);
  c03 = wmma16(a0, b3, c03);
  c10 = wmma16(a1, b0, c10);
  c11 = wmma16(a1, b1, c11);
  c12 = wmma16(a1, b2, c12);
  c13 = wmma16(a1, b3, c13);
#pragma unroll
  for (int r = 0; r < 8; ++r) {
    size_t ro0 = ((size_t)n * PP + i0 + hi * 8 + r) * PP + j0 + l15;
    size_t ro1 = ((size_t)n * PP + i0 + 16 + hi * 8 + r) * PP + j0 + l15;
    float v;
    v = c00[r]; corr16[ro0 + 0]  = (_Float16)(v > 0.f ? v : 0.f);
    v = c01[r]; corr16[ro0 + 16] = (_Float16)(v > 0.f ? v : 0.f);
    v = c02[r]; corr16[ro0 + 32] = (_Float16)(v > 0.f ? v : 0.f);
    v = c03[r]; corr16[ro0 + 48] = (_Float16)(v > 0.f ? v : 0.f);
    v = c10[r]; corr16[ro1 + 0]  = (_Float16)(v > 0.f ? v : 0.f);
    v = c11[r]; corr16[ro1 + 16] = (_Float16)(v > 0.f ? v : 0.f);
    v = c12[r]; corr16[ro1 + 32] = (_Float16)(v > 0.f ? v : 0.f);
    v = c13[r]; corr16[ro1 + 48] = (_Float16)(v > 0.f ? v : 0.f);
  }
}

// rowsum[n][i] = sum_j corr. One wave per row. grid (128, N).
__global__ void k_corr_rowsum(const _Float16* __restrict__ corr16,
                              float* __restrict__ rowsum) {
  int n = blockIdx.y;
  int wave = threadIdx.x >> 5, lane = threadIdx.x & 31;
  int i = blockIdx.x * 8 + wave;
  const _Float16* row = corr16 + ((size_t)n * PP + i) * PP;
  float s = 0.f;
  for (int j = lane; j < PP; j += 32) s += (float)row[j];
  for (int m = 16; m > 0; m >>= 1) s += __shfl_xor(s, m, 32);
  if (lane == 0) rowsum[n * PP + i] = s;
}

// inter[n][i][c] = sum_j corr[i][j]*camf[c][j] via WMMA, 1-deep prefetch.
// grid (16, N).
__global__ void __launch_bounds__(256) k_inter_wmma(
    const _Float16* __restrict__ corr16, const _Float16* __restrict__ nc16,
    float* __restrict__ inter) {
  int n = blockIdx.y;
  int wave = threadIdx.x >> 5, lane = threadIdx.x & 31;
  int l15 = lane & 15, hi = lane >> 4;
  int i0 = (blockIdx.x * 4 + (wave & 3)) * 16;
  int c0 = (wave >> 2) * 16;
  const _Float16* A = corr16 + ((size_t)n * PP + i0 + l15) * PP;
  const _Float16* B = nc16 + ((size_t)n * CP + c0 + l15) * PP + hi * 16;
  v8f acc = {};
  v16h a = load_afrag(A, 0, hi);
  v16h b = *(const v16h*)(B);
  for (int k0 = 32; k0 < PP; k0 += 32) {
    v16h na = load_afrag(A, k0, hi);
    v16h nb = *(const v16h*)(B + k0);
    acc = wmma16(a, b, acc);
    a = na; b = nb;
  }
  acc = wmma16(a, b, acc);
#pragma unroll
  for (int r = 0; r < 8; ++r)
    inter[((size_t)n * PP + i0 + hi * 8 + r) * CP + c0 + l15] = acc[r];
}

// miou + argmax -> belonging (first-max tie break like jnp.argmax). grid (4, N).
__global__ void k_miou_belong(const float* __restrict__ inter,
                              const float* __restrict__ rowsum,
                              const float* __restrict__ camsum,
                              int* __restrict__ bel1) {
  int n = blockIdx.y;
  int i = blockIdx.x * 256 + threadIdx.x;
  float rs = rowsum[n * PP + i];
  float best = -1e30f;
  int bc = 0;
  for (int c = 0; c < NC; ++c) {
    float iv = inter[((size_t)n * PP + i) * CP + c];
    float un = rs + camsum[n * NC + c] - iv;
    float m = iv / un;
    if (c == 0) m *= 0.5f;
    if (m > best) { best = m; bc = c; }
  }
  bel1[n * PP + i] = bc;
}

// proto[n][c][d] = sum_{p: belong==c, valid} hie[n][d][p] / 1024.
// grid (DH/8, N), block 256 = (32 px, 8 d), LDS-atomic class bins.
__global__ void k_proto(const float* __restrict__ hie,
                        const int* __restrict__ bel1,
                        const float* __restrict__ vmask,
                        float* __restrict__ proto) {
  int n = blockIdx.y, d0 = blockIdx.x * 8;
  int px = threadIdx.x & 31, dd = threadIdx.x >> 5;
  __shared__ float acc[NC][8];
  __shared__ float vm[NC];
  int tid = threadIdx.x;
  if (tid < NC) vm[tid] = vmask[n * NC + tid];
  for (int i = tid; i < NC * 8; i += 256) acc[i >> 3][i & 7] = 0.f;
  __syncthreads();
  for (int p0 = 0; p0 < PP; p0 += 32) {
    int p = p0 + px;
    int c = bel1[n * PP + p];
    float v = hie[((size_t)n * DH + d0 + dd) * PP + p];
    if (vm[c] > 0.f) atomicAdd(&acc[c][dd], v);
  }
  __syncthreads();
  for (int i = tid; i < NC * 8; i += 256)
    proto[((size_t)n * CP + (i >> 3)) * DH + d0 + (i & 7)] =
        acc[i >> 3][i & 7] * (1.f / (float)PP);
}

// protoN16 = l2-normalized prototypes (eps 1e-8), rows >= 21 zeroed. grid (32, N).
__global__ void k_protoN(const float* __restrict__ proto,
                         _Float16* __restrict__ protoN) {
  int c = blockIdx.x, n = blockIdx.y;
  size_t base = ((size_t)n * CP + c) * DH;
  if (c >= NC) {
    for (int d = threadIdx.x; d < DH; d += 256) protoN[base + d] = (_Float16)0.f;
    return;
  }
  float ssq = 0.f;
  for (int d = threadIdx.x; d < DH; d += 256) {
    float v = proto[base + d];
    ssq += v * v;
  }
  __shared__ float red[256];
  float tot = blk_sum(ssq, red);
  float inv = 1.f / (sqrtf(tot) + 1e-8f);
  for (int d = threadIdx.x; d < DH; d += 256)
    protoN[base + d] = (_Float16)(proto[base + d] * inv);
}

// IS[n][c][p] = relu(protoN . hieN) via WMMA, 1-deep prefetch. grid (16, N).
__global__ void __launch_bounds__(256) k_react_wmma(
    const _Float16* __restrict__ protoN, const _Float16* __restrict__ hieNT,
    float* __restrict__ IS) {
  int n = blockIdx.y;
  int wave = threadIdx.x >> 5, lane = threadIdx.x & 31;
  int l15 = lane & 15, hi = lane >> 4;
  int p0 = (blockIdx.x * 4 + (wave & 3)) * 16;
  int c0 = (wave >> 2) * 16;
  const _Float16* A = protoN + ((size_t)n * CP + c0 + l15) * DH;
  const _Float16* B = hieNT + ((size_t)n * PP + p0 + l15) * DH + hi * 16;
  v8f acc = {};
  v16h a = load_afrag(A, 0, hi);
  v16h b = *(const v16h*)(B);
  for (int k0 = 32; k0 < DH; k0 += 32) {
    v16h na = load_afrag(A, k0, hi);
    v16h nb = *(const v16h*)(B + k0);
    acc = wmma16(a, b, acc);
    a = na; b = nb;
  }
  acc = wmma16(a, b, acc);
#pragma unroll
  for (int r = 0; r < 8; ++r) {
    float v = acc[r];
    IS[((size_t)n * CP + c0 + hi * 8 + r) * PP + p0 + l15] = v > 0.f ? v : 0.f;
  }
}

// Per-(n,c) max-normalize IS in place (eps 1e-5). grid (21, N).
__global__ void k_IS_norm(float* __restrict__ IS) {
  int c = blockIdx.x, n = blockIdx.y;
  float* row = IS + ((size_t)n * CP + c) * PP;
  float m = 0.f;
  for (int p = threadIdx.x; p < PP; p += 256) m = fmaxf(m, row[p]);
  __shared__ float red[256];
  float tm = blk_max(m, red);
  float inv = 1.f / (tm + 1e-5f);
  for (int p = threadIdx.x; p < PP; p += 256) row[p] *= inv;
}

// belonging2 = argmax_c IS. grid (4, N).
__global__ void k_belong2(const float* __restrict__ IS, int* __restrict__ bel2) {
  int n = blockIdx.y;
  int p = blockIdx.x * 256 + threadIdx.x;
  float best = -1e30f;
  int bc = 0;
  for (int c = 0; c < NC; ++c) {
    float v = IS[((size_t)n * CP + c) * PP + p];
    if (v > best) { best = v; bc = c; }
  }
  bel2[n * PP + p] = bc;
}

// ---------------------------------------------------------------------------
// Per-(n,c) 2-cluster k-means over X=hie^T (1024x768), entirely in one block.
// Centers live in LDS; assign pass = wave-per-pixel dot products; update pass
// = thread-per-dim masked accumulation. grid (21, N), block 256.
// ---------------------------------------------------------------------------
__global__ void __launch_bounds__(256) k_kmeans(
    const float* __restrict__ X, const float* __restrict__ ssHie,
    const int* __restrict__ bel2, const float* __restrict__ vmask,
    float* __restrict__ centers, float* __restrict__ camkm) {
  int c = blockIdx.x, n = blockIdx.y;
  int tid = threadIdx.x, wave = tid >> 5, lane = tid & 31;
  __shared__ float cen[2][DH];
  __shared__ float nacc[2][DH];
  __shared__ unsigned char mk[PP];
  __shared__ unsigned char asg[PP];
  __shared__ float red[256];
  __shared__ float wbv[8];
  __shared__ int wbp[8];
  __shared__ int shp;
  const float* Xn = X + (size_t)n * PP * DH;
  float label = vmask[n * NC + c];
  float cl = 0.f;
  for (int p = tid; p < PP; p += 256) {
    unsigned char m = (label > 0.f && bel2[n * PP + p] == c) ? 1 : 0;
    mk[p] = m;
    cl += (float)m;
  }
  float cnt = blk_sum(cl, red);
  size_t cbase = (size_t)(n * NC + c) * 2 * DH;
  if (cnt < 2.f) {  // invalid class: zero centers & cam, matching reference
    for (int d = tid; d < 2 * DH; d += 256) centers[cbase + d] = 0.f;
    for (int p = tid; p < PP; p += 256)
      camkm[(size_t)(n * NC + c) * PP + p] = 0.f;
    return;
  }
  // c0 = masked mean
  for (int d = tid; d < DH; d += 256) {
    float a = 0.f;
    for (int p = 0; p < PP; ++p) a += (float)mk[p] * Xn[(size_t)p * DH + d];
    cen[0][d] = a / cnt;
  }
  __syncthreads();
  float sq = 0.f;
  for (int d = tid; d < DH; d += 256) { float v = cen[0][d]; sq += v * v; }
  float inv0 = 1.f / (sqrtf(blk_sum(sq, red)) + 1e-8f);
  // c1 = farthest masked point from c0 (cosine distance), first-index ties
  float bv = -1e30f;
  int bp = 0;
  for (int it = 0; it < 128; ++it) {
    int p = it * 8 + wave;
    float dot = 0.f;
    for (int d = lane; d < DH; d += 32) dot += Xn[(size_t)p * DH + d] * cen[0][d];
    for (int m = 16; m > 0; m >>= 1) dot += __shfl_xor(dot, m, 32);
    float invX = 1.f / (sqrtf(ssHie[n * PP + p]) + 1e-8f);
    float val = mk[p] ? (1.f - dot * invX * inv0) : -1e30f;
    if (val > bv || (val == bv && p < bp)) { bv = val; bp = p; }
  }
  if (lane == 0) { wbv[wave] = bv; wbp[wave] = bp; }
  __syncthreads();
  if (tid == 0) {
    float b = wbv[0]; int q = wbp[0];
    for (int w = 1; w < 8; ++w)
      if (wbv[w] > b || (wbv[w] == b && wbp[w] < q)) { b = wbv[w]; q = wbp[w]; }
    shp = q;
  }
  __syncthreads();
  int pstar = shp;
  for (int d = tid; d < DH; d += 256) cen[1][d] = Xn[(size_t)pstar * DH + d];
  __syncthreads();
  // 10 Lloyd iterations
  for (int iter = 0; iter < 10; ++iter) {
    float s0 = 0.f, s1 = 0.f;
    for (int d = tid; d < DH; d += 256) {
      float a = cen[0][d], b = cen[1][d];
      s0 += a * a; s1 += b * b;
    }
    float ic0 = 1.f / (sqrtf(blk_sum(s0, red)) + 1e-8f);
    float ic1 = 1.f / (sqrtf(blk_sum(s1, red)) + 1e-8f);
    float c0l = 0.f, c1l = 0.f;
    for (int it2 = 0; it2 < 128; ++it2) {
      int p = it2 * 8 + wave;
      float dA = 0.f, dB = 0.f;
      for (int d = lane; d < DH; d += 32) {
        float x = Xn[(size_t)p * DH + d];
        dA += x * cen[0][d];
        dB += x * cen[1][d];
      }
      for (int m = 16; m > 0; m >>= 1) {
        dA += __shfl_xor(dA, m, 32);
        dB += __shfl_xor(dB, m, 32);
      }
      float invX = 1.f / (sqrtf(ssHie[n * PP + p]) + 1e-8f);
      unsigned char a = (dB * invX * ic1 > dA * invX * ic0) ? 1 : 0;
      if (lane == 0) {
        asg[p] = a;
        float mm = (float)mk[p];
        c0l += mm * (float)(1 - a);
        c1l += mm * (float)a;
      }
    }
    float cnt0 = blk_sum(c0l, red);
    float cnt1 = blk_sum(c1l, red);
    for (int d = tid; d < DH; d += 256) {
      float a0 = 0.f, a1 = 0.f;
      for (int p = 0; p < PP; ++p) {
        float mm = (float)mk[p];
        float aa = (float)asg[p];
        float x = Xn[(size_t)p * DH + d];
        float w1 = mm * aa;
        a0 += (mm - w1) * x;
        a1 += w1 * x;
      }
      nacc[0][d] = a0;
      nacc[1][d] = a1;
    }
    __syncthreads();
    for (int d = tid; d < DH; d += 256) {
      if (cnt0 > 0.f) cen[0][d] = nacc[0][d] / cnt0;
      if (cnt1 > 0.f) cen[1][d] = nacc[1][d] / cnt1;
    }
    __syncthreads();
  }
  // final cam = mean over the 2 clusters of Xn . cenn
  float s0 = 0.f, s1 = 0.f;
  for (int d = tid; d < DH; d += 256) {
    float a = cen[0][d], b = cen[1][d];
    s0 += a * a; s1 += b * b;
  }
  float ic0 = 1.f / (sqrtf(blk_sum(s0, red)) + 1e-8f);
  float ic1 = 1.f / (sqrtf(blk_sum(s1, red)) + 1e-8f);
  for (int it2 = 0; it2 < 128; ++it2) {
    int p = it2 * 8 + wave;
    float dA = 0.f, dB = 0.f;
    for (int d = lane; d < DH; d += 32) {
      float x = Xn[(size_t)p * DH + d];
      dA += x * cen[0][d];
      dB += x * cen[1][d];
    }
    for (int m = 16; m > 0; m >>= 1) {
      dA += __shfl_xor(dA, m, 32);
      dB += __shfl_xor(dB, m, 32);
    }
    if (lane == 0) {
      float invX = 1.f / (sqrtf(ssHie[n * PP + p]) + 1e-8f);
      camkm[(size_t)(n * NC + c) * PP + p] = 0.5f * (dA * ic0 + dB * ic1) * invX;
    }
  }
  for (int d = tid; d < DH; d += 256) {
    centers[cbase + d] = cen[0][d];
    centers[cbase + DH + d] = cen[1][d];
  }
}

// all_cam = cams*label, max-normalized per (n,c) -> allN (reusing IS buffer).
__global__ void k_allcamA(const float* __restrict__ camkm,
                          const float* __restrict__ vmask,
                          float* __restrict__ allN) {
  int c = blockIdx.x, n = blockIdx.y;
  const float* row = camkm + (size_t)(n * NC + c) * PP;
  float label = vmask[n * NC + c];
  float m = -1e30f;
  for (int p = threadIdx.x; p < PP; p += 256) m = fmaxf(m, row[p] * label);
  __shared__ float red[256];
  float tm = blk_max(m, red);
  float inv = 1.f / (tm + 1e-5f);
  for (int p = threadIdx.x; p < PP; p += 256)
    allN[((size_t)n * CP + c) * PP + p] = row[p] * label * inv;
}

// proto_cam = concat(1 - max_{c>=1} allN, allN[1:]). grid (4, N).
__global__ void k_protocamB(const float* __restrict__ allN,
                            float* __restrict__ pcOut) {
  int n = blockIdx.y;
  int p = blockIdx.x * 256 + threadIdx.x;
  float mx = -1e30f;
  for (int c = 1; c < NC; ++c) {
    float v = allN[((size_t)n * CP + c) * PP + p];
    pcOut[((size_t)n * NC + c) * PP + p] = v;
    mx = fmaxf(mx, v);
  }
  pcOut[(size_t)n * NC * PP + p] = 1.f - mx;
}

// ---------------------------------------------------------------------------
// Sinkhorn per (n,c): K (1024x2) built on the fly from the two centers mapped
// by the reference's reshape (j = k*21 + c -> class j/2, cluster j%2),
// iterated entirely in LDS. grid (21, N).
// ---------------------------------------------------------------------------
__global__ void __launch_bounds__(256) k_sinkhorn(
    const float* __restrict__ X, const float* __restrict__ ssHie,
    const float* __restrict__ centers, float* __restrict__ Tout) {
  int c = blockIdx.x, n = blockIdx.y;
  int tid = threadIdx.x, wave = tid >> 5, lane = tid & 31;
  __shared__ float KL0[PP], KL1[PP], rr[PP];
  __shared__ float cA[DH], cB[DH];
  __shared__ float red[256];
  int j0 = c, j1 = NC + c;
  const float* CA = centers + ((size_t)(n * NC + (j0 >> 1)) * 2 + (j0 & 1)) * DH;
  const float* CB = centers + ((size_t)(n * NC + (j1 >> 1)) * 2 + (j1 & 1)) * DH;
  float s0 = 0.f, s1 = 0.f;
  for (int d = tid; d < DH; d += 256) {
    float a = CA[d], b = CB[d];
    cA[d] = a; cB[d] = b;
    s0 += a * a; s1 += b * b;
  }
  float icA = 1.f / (sqrtf(blk_sum(s0, red)) + 1e-5f);
  float icB = 1.f / (sqrtf(blk_sum(s1, red)) + 1e-5f);
  const float* Xn = X + (size_t)n * PP * DH;
  for (int it = 0; it < 128; ++it) {
    int p = it * 8 + wave;
    float dA = 0.f, dB = 0.f;
    for (int d = lane; d < DH; d += 32) {
      float x = Xn[(size_t)p * DH + d];
      dA += x * cA[d];
      dB += x * cB[d];
    }
    for (int m = 16; m > 0; m >>= 1) {
      dA += __shfl_xor(dA, m, 32);
      dB += __shfl_xor(dB, m, 32);
    }
    if (lane == 0) {
      float invF = 1.f / (sqrtf(ssHie[n * PP + p]) + 1e-5f);
      KL0[p] = __expf(10.f * (dA * invF * icA - 1.f));
      KL1[p] = __expf(10.f * (dB * invF * icB - 1.f));
      rr[p] = 1.f;
    }
  }
  __syncthreads();
  float cv0 = 1.f, cv1 = 1.f;
  const float u = 1.f / (float)PP, v = 0.5f;
  for (int it = 0; it < 100; ++it) {
    float errp = 0.f, S0p = 0.f, S1p = 0.f;
    float rn[4];
#pragma unroll
    for (int q = 0; q < 4; ++q) {
      int p = tid + 256 * q;
      float denom = KL0[p] * cv0 + KL1[p] * cv1;
      float r = u / denom;
      errp += fabsf(r - rr[p]);
      rn[q] = r;
      S0p += KL0[p] * r;
      S1p += KL1[p] * r;
    }
#pragma unroll
    for (int q = 0; q < 4; ++q) rr[tid + 256 * q] = rn[q];
    float S0 = blk_sum(S0p, red);
    float S1 = blk_sum(S1p, red);
    float err = blk_sum(errp, red) * (1.f / (float)PP);
    cv0 = v / S0;
    cv1 = v / S1;
    if (err < 0.01f) break;
  }
#pragma unroll
  for (int q = 0; q < 4; ++q) {
    int p = tid + 256 * q;
    size_t o = ((size_t)(n * NC + c) * PP + p) * 2;
    Tout[o] = rr[p] * cv0 * KL0[p];
    Tout[o + 1] = rr[p] * cv1 * KL1[p];
  }
}

// ---------------------------------------------------------------------------
extern "C" void kernel_launch(void* const* d_in, const int* in_sizes, int n_in,
                              void* d_out, int out_size, void* d_ws,
                              size_t ws_size, hipStream_t stream) {
  (void)in_sizes; (void)n_in; (void)out_size; (void)ws_size;
  const float* sem = (const float*)d_in[0];
  const float* hie = (const float*)d_in[1];
  const float* wcls = (const float*)d_in[2];
  const float* vmask = (const float*)d_in[3];
  float* out = (float*)d_out;
  float* score = out;                           // (8,20,1,1)
  float* ncOut = out + 160;                     // (8,21,32,32)
  float* pcOut = ncOut + (size_t)NB * NC * PP;  // (8,21,32,32)
  float* TOut = pcOut + (size_t)NB * NC * PP;   // (8,21,1024,2)

  char* W = (char*)d_ws;
  size_t off = 0;
  auto alloc = [&](size_t b) -> void* {
    void* p = W + off;
    off += (b + 255) & ~(size_t)255;
    return p;
  };
  _Float16* fsT = (_Float16*)alloc((size_t)NB * PP * DS * 2);   // normalized sem^T f16
  _Float16* hieNT = (_Float16*)alloc((size_t)NB * PP * DH * 2); // normalized hie^T f16
  float* hieT = (float*)alloc((size_t)NB * PP * DH * 4);        // raw hie^T f32 (X)
  float* ssSem = (float*)alloc((size_t)NB * PP * 4);
  float* ssHie = (float*)alloc((size_t)NB * PP * 4);
  float* cam = (float*)alloc((size_t)NB * 20 * PP * 4);
  float* camMax = (float*)alloc((size_t)NB * 20 * 4);
  _Float16* nc16 = (_Float16*)alloc((size_t)NB * CP * PP * 2);
  float* camsum = (float*)alloc((size_t)NB * NC * 4);
  _Float16* corr16 = (_Float16*)alloc((size_t)NB * PP * PP * 2);
  float* rowsum = (float*)alloc((size_t)NB * PP * 4);
  float* inter = (float*)alloc((size_t)NB * PP * CP * 4);
  int* bel1 = (int*)alloc((size_t)NB * PP * 4);
  float* proto = (float*)alloc((size_t)NB * CP * DH * 4);
  _Float16* protoN = (_Float16*)alloc((size_t)NB * CP * DH * 2);
  float* ISb = (float*)alloc((size_t)NB * CP * PP * 4);
  int* bel2 = (int*)alloc((size_t)NB * PP * 4);
  float* cent = (float*)alloc((size_t)NB * NC * 2 * DH * 4);
  float* camkm = (float*)alloc((size_t)NB * NC * PP * 4);

  k_sumsq<<<dim3(32, NB), 256, 0, stream>>>(sem, ssSem, DS);
  k_sumsq<<<dim3(32, NB), 256, 0, stream>>>(hie, ssHie, DH);
  k_transpose_norm<<<dim3(DS / 64, 32, NB), 256, 0, stream>>>(
      sem, ssSem, fsT, nullptr, DS, 1e-5f);
  k_transpose_norm<<<dim3(DH / 64, 32, NB), 256, 0, stream>>>(
      hie, ssHie, hieNT, hieT, DH, 1e-8f);
  k_cam<<<dim3(16, NB), 256, 0, stream>>>(sem, wcls, cam);
  k_cam_max_score<<<dim3(20, NB), 256, 0, stream>>>(cam, score, camMax);
  k_normcam<<<dim3(4, NB), 256, 0, stream>>>(cam, camMax, vmask, ncOut, nc16);
  k_camsum<<<dim3(NC, NB), 256, 0, stream>>>(ncOut, camsum);
  k_corr_wmma<<<dim3(16, 4, NB), 256, 0, stream>>>(fsT, corr16);
  k_corr_rowsum<<<dim3(128, NB), 256, 0, stream>>>(corr16, rowsum);
  k_inter_wmma<<<dim3(16, NB), 256, 0, stream>>>(corr16, nc16, inter);
  k_miou_belong<<<dim3(4, NB), 256, 0, stream>>>(inter, rowsum, camsum, bel1);
  k_proto<<<dim3(DH / 8, NB), 256, 0, stream>>>(hie, bel1, vmask, proto);
  k_protoN<<<dim3(CP, NB), 256, 0, stream>>>(proto, protoN);
  k_react_wmma<<<dim3(16, NB), 256, 0, stream>>>(protoN, hieNT, ISb);
  k_IS_norm<<<dim3(NC, NB), 256, 0, stream>>>(ISb);
  k_belong2<<<dim3(4, NB), 256, 0, stream>>>(ISb, bel2);
  k_kmeans<<<dim3(NC, NB), 256, 0, stream>>>(hieT, ssHie, bel2, vmask, cent,
                                             camkm);
  k_allcamA<<<dim3(NC, NB), 256, 0, stream>>>(camkm, vmask, ISb);
  k_protocamB<<<dim3(4, NB), 256, 0, stream>>>(ISb, pcOut);
  k_sinkhorn<<<dim3(NC, NB), 256, 0, stream>>>(hieT, ssHie, cent, TOut);
}
